// GCNComm_33079838114378
// MI455X (gfx1250) — matrix operand order repaired and checked by reading, not compiled, not run on hardware
//
#include <hip/hip_runtime.h>
#include <hip/hip_bf16.h>
#include <cstdint>

// Problem constants (match reference)
#define BB 4
#define NN 4096
#define DD 64

typedef __attribute__((ext_vector_type(16))) __bf16 v16bf;
typedef __attribute__((ext_vector_type(8)))  float  v8f;

union Frag16 {
    uint4 u[2];   // two b128 loads
    v16bf v;      // 16 bf16 = 8 VGPRs
};

__device__ __forceinline__ unsigned short f2bf(float f) {
    uint32_t x = __builtin_bit_cast(uint32_t, f);
    uint32_t r = x + 0x7FFFu + ((x >> 16) & 1u);   // round-to-nearest-even
    return (unsigned short)(r >> 16);
}

__device__ __forceinline__ float rnd01(uint32_t x) {
    x ^= x >> 17; x *= 0xed5ad4bbu;
    x ^= x >> 11; x *= 0xac4c1b51u;
    x ^= x >> 15; x *= 0x31848babu;
    x ^= x >> 14;
    return (float)(x >> 8) * (1.0f / 16777216.0f);
}

// ---------------------------------------------------------------------------
// 1) dinv[b][i] = rsqrt(1 + #nonzero offdiag in row i)  (self-loop forced to 1)
// ---------------------------------------------------------------------------
__global__ __launch_bounds__(256)
void dinv_kernel(const int* __restrict__ adj, float* __restrict__ dinv) {
    __shared__ int red[256];
    const int blk = blockIdx.x;                 // B*N blocks
    const int i = blk & (NN - 1);
    const int* row = adj + (size_t)blk * NN;
    int cnt = 0;
    #pragma unroll 4
    for (int j = threadIdx.x; j < NN; j += 256)
        cnt += (row[j] != 0 && j != i) ? 1 : 0;
    red[threadIdx.x] = cnt;
    __syncthreads();
    for (int s = 128; s > 0; s >>= 1) {
        if (threadIdx.x < s) red[threadIdx.x] += red[threadIdx.x + s];
        __syncthreads();
    }
    if (threadIdx.x == 0)
        dinv[blk] = rsqrtf((float)(red[0] + 1));
}

// ---------------------------------------------------------------------------
// 2) Ahat[b][i][j] = (adj!=0 || i==j) ? dinv_i*dinv_j : 0   (bf16, 8 elems/thr)
// ---------------------------------------------------------------------------
__global__ __launch_bounds__(256)
void ahat_kernel(const int* __restrict__ adj, const float* __restrict__ dinv,
                 unsigned short* __restrict__ Ahat) {
    const int g = blockIdx.x * 256 + threadIdx.x;        // granule of 8 j's
    const int j0 = (g & (NN / 8 - 1)) * 8;
    const int i  = (g / (NN / 8)) & (NN - 1);
    const int b  = g / ((NN / 8) * NN);
    const size_t base = ((size_t)(b * NN + i)) * NN + j0;
    const int4 a0 = *(const int4*)(adj + base);
    const int4 a1 = *(const int4*)(adj + base + 4);
    const float di = dinv[b * NN + i];
    const float* dj = dinv + b * NN + j0;
    int av[8] = {a0.x, a0.y, a0.z, a0.w, a1.x, a1.y, a1.z, a1.w};
    unsigned short o[8];
    #pragma unroll
    for (int t = 0; t < 8; ++t) {
        const bool on = (av[t] != 0) || (j0 + t == i);
        o[t] = f2bf(on ? di * dj[t] : 0.0f);
    }
    uint4 pack;
    pack.x = (uint32_t)o[0] | ((uint32_t)o[1] << 16);
    pack.y = (uint32_t)o[2] | ((uint32_t)o[3] << 16);
    pack.z = (uint32_t)o[4] | ((uint32_t)o[5] << 16);
    pack.w = (uint32_t)o[6] | ((uint32_t)o[7] << 16);
    *(uint4*)(Ahat + base) = pack;
}

// ---------------------------------------------------------------------------
// 3) T[b][d][j] = bf16( sum_k H[b][j][k] * W[k][d] )   (transposed, d-major)
// ---------------------------------------------------------------------------
__global__ __launch_bounds__(256)
void xw_kernel(const float* __restrict__ H, const float* __restrict__ W,
               unsigned short* __restrict__ T) {
    __shared__ float Wl[DD * DD];
    for (int t = threadIdx.x; t < DD * DD; t += 256) Wl[t] = W[t];
    __syncthreads();
    const int blk = blockIdx.x;                  // B * N/4 blocks
    const int b  = blk / (NN / 4);
    const int j  = (blk % (NN / 4)) * 4 + (threadIdx.x >> 6);
    const int d  = threadIdx.x & 63;
    const float* hrow = H + ((size_t)(b * NN + j)) * DD;
    float s = 0.0f;
    #pragma unroll
    for (int k = 0; k < DD; ++k) s += hrow[k] * Wl[k * DD + d];
    T[((size_t)(b * DD + d)) * NN + j] = f2bf(s);
}

// ---------------------------------------------------------------------------
// 4) Out[b][i][d] = sum_j Ahat[b][i][j] * T[b][d][j] + bias[d]  (+ELU+dropout)
//    Wave tile: 16 rows x 64 cols, K-loop in steps of 32, bf16 WMMA, f32 acc.
// ---------------------------------------------------------------------------
__global__ __launch_bounds__(128)
void spmm_wmma(const unsigned short* __restrict__ Ahat,   // [B][N][N] bf16
               const unsigned short* __restrict__ T,      // [B][64][N] bf16
               const float* __restrict__ bias,            // [64]
               float* __restrict__ Hout,                  // [B][N][64] f32
               int act, uint32_t seed) {
    const int wave = threadIdx.x >> 5;
    const int lane = threadIdx.x & 31;
    const int half = lane >> 4;
    const int l16  = lane & 15;
    const int blk  = blockIdx.x;                 // B * N/64 blocks
    const int b    = blk / (NN / 64);
    const int row0 = (blk % (NN / 64)) * 64 + wave * 16;

    const unsigned short* arow = Ahat + ((size_t)(b * NN + row0 + l16)) * NN;
    const unsigned short* tb   = T + (size_t)b * DD * NN;
    const unsigned short* tc0  = tb + ((size_t)(0 * 16 + l16)) * NN;
    const unsigned short* tc1  = tb + ((size_t)(1 * 16 + l16)) * NN;
    const unsigned short* tc2  = tb + ((size_t)(2 * 16 + l16)) * NN;
    const unsigned short* tc3  = tb + ((size_t)(3 * 16 + l16)) * NN;

    v8f acc0 = {}, acc1 = {}, acc2 = {}, acc3 = {};

    for (int k0 = 0; k0 < NN; k0 += 32) {
        // A fragment (16x32 bf16): k = k0 + half*16 + [0,16), contiguous
        Frag16 fa;
        fa.u[0] = *(const uint4*)(arow + k0 + half * 16);
        fa.u[1] = *(const uint4*)(arow + k0 + half * 16 + 8);
        // B fragments (32x16 bf16): k = k0 + half*16 + [0,16), contiguous
        Frag16 fb0, fb1, fb2, fb3;
        const int kb = k0 + half * 16;
        fb0.u[0] = *(const uint4*)(tc0 + kb);     fb0.u[1] = *(const uint4*)(tc0 + kb + 8);
        fb1.u[0] = *(const uint4*)(tc1 + kb);     fb1.u[1] = *(const uint4*)(tc1 + kb + 8);
        fb2.u[0] = *(const uint4*)(tc2 + kb);     fb2.u[1] = *(const uint4*)(tc2 + kb + 8);
        fb3.u[0] = *(const uint4*)(tc3 + kb);     fb3.u[1] = *(const uint4*)(tc3 + kb + 8);

        acc0 = __builtin_amdgcn_wmma_f32_16x16x32_bf16(false, fa.v, false, fb0.v,
                                                       (short)0, acc0, false, false);
        acc1 = __builtin_amdgcn_wmma_f32_16x16x32_bf16(false, fa.v, false, fb1.v,
                                                       (short)0, acc1, false, false);
        acc2 = __builtin_amdgcn_wmma_f32_16x16x32_bf16(false, fa.v, false, fb2.v,
                                                       (short)0, acc2, false, false);
        acc3 = __builtin_amdgcn_wmma_f32_16x16x32_bf16(false, fa.v, false, fb3.v,
                                                       (short)0, acc3, false, false);
    }

    // Epilogue: C/D layout => lane L: col = n*16 + L%16, row = r + 8*(L/16)
    v8f accs[4] = {acc0, acc1, acc2, acc3};
    #pragma unroll
    for (int n = 0; n < 4; ++n) {
        const int col = n * 16 + l16;
        const float bi = bias[col];
        #pragma unroll
        for (int r = 0; r < 8; ++r) {
            const int row = row0 + r + 8 * half;
            float v = accs[n][r] + bi;
            if (act) {
                v = (v > 0.0f) ? v : (__expf(v) - 1.0f);            // ELU
                const uint32_t idx = (uint32_t)(((b * NN + row) * DD) + col);
                const float u = rnd01(idx * 0x9E3779B9u ^ seed);
                v = (u < 0.8f) ? v * 1.25f : 0.0f;                  // dropout p=0.2
            }
            Hout[((size_t)(b * NN + row)) * DD + col] = v;
        }
    }
}

// ---------------------------------------------------------------------------
extern "C" void kernel_launch(void* const* d_in, const int* in_sizes, int n_in,
                              void* d_out, int out_size, void* d_ws, size_t ws_size,
                              hipStream_t stream) {
    (void)in_sizes; (void)n_in; (void)out_size; (void)ws_size;
    const float* x  = (const float*)d_in[0];
    const int*  adj = (const int*)d_in[1];
    const float* W1 = (const float*)d_in[2];
    const float* b1 = (const float*)d_in[3];
    const float* W2 = (const float*)d_in[4];
    const float* b2 = (const float*)d_in[5];
    const float* W3 = (const float*)d_in[6];
    const float* b3 = (const float*)d_in[7];
    float* out = (float*)d_out;

    // Workspace layout
    char* ws = (char*)d_ws;
    const size_t szAhat = (size_t)BB * NN * NN * 2;        // 128 MB  bf16
    const size_t szDinv = (size_t)BB * NN * 4;             // 64 KB   f32
    const size_t szT    = (size_t)BB * DD * NN * 2;        // 2 MB    bf16
    unsigned short* Ahat = (unsigned short*)ws;
    float*          dinv = (float*)(ws + szAhat);
    unsigned short* T    = (unsigned short*)(ws + szAhat + szDinv);
    float*          H    = (float*)(ws + szAhat + szDinv + szT);  // 4 MB f32

    // 1) degrees -> dinv
    dinv_kernel<<<BB * NN, 256, 0, stream>>>(adj, dinv);
    // 2) normalized adjacency in bf16
    ahat_kernel<<<(BB * NN * (NN / 8)) / 256, 256, 0, stream>>>(adj, dinv, Ahat);

    // Layer 1
    xw_kernel<<<BB * (NN / 4), 256, 0, stream>>>(x, W1, T);
    spmm_wmma<<<BB * (NN / 64), 128, 0, stream>>>(Ahat, T, b1, H, 1, 0x2a000001u);
    // Layer 2
    xw_kernel<<<BB * (NN / 4), 256, 0, stream>>>(H, W2, T);
    spmm_wmma<<<BB * (NN / 64), 128, 0, stream>>>(Ahat, T, b2, H, 1, 0x2a000002u);
    // Layer 3 (no activation / dropout)
    xw_kernel<<<BB * (NN / 4), 256, 0, stream>>>(H, W3, T);
    spmm_wmma<<<BB * (NN / 64), 128, 0, stream>>>(Ahat, T, b3, out, 0, 0u);
}
